// ParT_10565619548497
// MI455X (gfx1250) — matrix operand order repaired and checked by reading, not compiled, not run on hardware
//
#include <hip/hip_runtime.h>
#include <hip/hip_bf16.h>
#include <math.h>

// Model dims
#define B_  32
#define N_  256
#define D_  512
#define H_  8
#define DH_ 64
#define F_  2048
#define L_  8
#define LC_ 2
#define SCALE_ 0.125f        // 1/sqrt(64)
#define NEG_  (-1e9f)

typedef __attribute__((ext_vector_type(16))) __bf16 v16bf;
typedef __attribute__((ext_vector_type(8)))  float  v8f;

union ABFrag { v16bf v; unsigned u[8]; };
union CFrag  { v8f v; float f[8]; };

__device__ __forceinline__ unsigned short f2bf(float f) {
  unsigned u = __float_as_uint(f);
  unsigned r = u + 0x7FFFu + ((u >> 16) & 1u);
  return (unsigned short)(r >> 16);
}
__device__ __forceinline__ float bf2f(unsigned short b) {
  return __uint_as_float(((unsigned)b) << 16);
}

__device__ __forceinline__ float wave_sum(float v) {
#pragma unroll
  for (int o = 16; o > 0; o >>= 1) v += __shfl_xor(v, o, 32);
  return v;
}
__device__ __forceinline__ float wave_max(float v) {
#pragma unroll
  for (int o = 16; o > 0; o >>= 1) v = fmaxf(v, __shfl_xor(v, o, 32));
  return v;
}

__device__ __forceinline__ v8f wmma_bf16(v16bf a, v16bf b, v8f c) {
  return __builtin_amdgcn_wmma_f32_16x16x32_bf16(false, a, false, b, (short)0, c, false, false);
}

// CDNA5 async global -> LDS (16B per lane). loff = byte offset into workgroup LDS.
__device__ __forceinline__ void async_g2l_b128(unsigned loff, unsigned long long gaddr) {
  asm volatile("global_load_async_to_lds_b128 %0, %1, off"
               :: "v"(loff), "v"(gaddr) : "memory");
}
__device__ __forceinline__ void wait_async0() {
#if __has_builtin(__builtin_amdgcn_s_wait_asynccnt)
  __builtin_amdgcn_s_wait_asynccnt(0);
#else
  asm volatile("s_wait_asynccnt 0" ::: "memory");
#endif
}

// A fragment (16x32 bf16): lanes0-15 M=lane K=0..15; lanes16-31 K=16..31
__device__ __forceinline__ v16bf ldsA_frag(const unsigned short* __restrict__ sA, int row0, int ld) {
  int lane = threadIdx.x & 31;
  int m = lane & 15, kh = (lane >> 4) << 4;
  ABFrag f;
#pragma unroll
  for (int e = 0; e < 8; ++e) {
    int k = (e << 1) + kh;
    f.u[e] = *(const unsigned*)(sA + (row0 + m) * ld + k);
  }
  return f.v;
}
// B fragment (32x16 bf16) from LDS tile stored TRANSPOSED [cols][32 k]
__device__ __forceinline__ v16bf ldsBt_frag(const unsigned short* __restrict__ sBt, int col0) {
  int lane = threadIdx.x & 31;
  int n = lane & 15, kb = (lane >> 4) << 4;
  ABFrag f;
#pragma unroll
  for (int e = 0; e < 8; ++e)
    f.u[e] = *(const unsigned*)(sBt + (col0 + n) * 32 + kb + (e << 1));
  return f.v;
}

template<bool BIAS, bool GELU_, bool RES, bool OUTBF>
__device__ __forceinline__ void gemm_epilogue(const CFrag& c, int row0, int col0,
    int M, int N, const float* __restrict__ bias, const float* __restrict__ resid,
    float* __restrict__ outF, unsigned short* __restrict__ outB)
{
  int lane = threadIdx.x & 31;
  int col = col0 + (lane & 15);
  int rb = row0 + ((lane >> 4) << 3);
  float bv = 0.f;
  if (BIAS) bv = bias[col];
#pragma unroll
  for (int r = 0; r < 8; ++r) {
    int row = rb + r;
    if (row >= M) continue;
    float xv = c.f[r] + bv;
    if (GELU_) xv = 0.5f * xv * (1.f + erff(xv * 0.70710678118f));
    if (RES)   xv += resid[(size_t)row * N + col];
    if (OUTBF) outB[(size_t)row * N + col] = f2bf(xv);
    else       outF[(size_t)row * N + col] = xv;
  }
}

// C[M,N] = A[M,K]@W[K,N] (+bias)(+gelu)(+resid), bf16 in, f32 accumulate.
// Block tile 128x128, 8 waves, each wave 32x64 (2x4 WMMA 16x16x32 frags).
// Double-buffered LDS; A staged via async global->LDS, B staged via regs + transpose.
template<bool BIAS, bool GELU_, bool RES, bool OUTBF>
__global__ __launch_bounds__(256) void gemm_bf16_wmma(
    const unsigned short* __restrict__ A, const unsigned short* __restrict__ W,
    const float* __restrict__ bias, const float* __restrict__ resid,
    float* __restrict__ outF, unsigned short* __restrict__ outB,
    int M, int N, int K)
{
  extern __shared__ unsigned short smem[];
  // layout (elements): sA0 [0,4096), sA1 [4096,8192), sBt0 [8192,12288), sBt1 [12288,16384)
  const int t = threadIdx.x, wave = t >> 5;
  const int tileM = blockIdx.x * 128, tileN = blockIdx.y * 128;
  const int wm = (wave >> 1) << 5, wn = (wave & 1) << 6;

  CFrag acc[2][4];
#pragma unroll
  for (int i = 0; i < 2; ++i)
#pragma unroll
    for (int j = 0; j < 4; ++j)
#pragma unroll
      for (int r = 0; r < 8; ++r) acc[i][j].f[r] = 0.f;

  auto stageA = [&](int buf, int kk) {   // 128x32 bf16, 512 x 16B async chunks
#pragma unroll
    for (int i = 0; i < 2; ++i) {
      int cidx = t + (i << 8);
      int row = cidx >> 2, seg = cidx & 3;
      int grow = tileM + row; if (grow >= M) grow = 0;  // clamp; stores row-guarded
      unsigned long long gp = (unsigned long long)(A + (size_t)grow * K + kk + (seg << 3));
      unsigned loff = (unsigned)(buf * 8192 + ((row << 5) + (seg << 3)) * 2);
      async_g2l_b128(loff, gp);
    }
  };
  auto loadB = [&](int kk, uint4* r) {   // 32x128 bf16 -> 2 uint4 per thread
#pragma unroll
    for (int i = 0; i < 2; ++i) {
      int c = t + (i << 8);
      int kr = c >> 4, c8 = (c & 15) << 3;
      r[i] = *(const uint4*)(W + (size_t)(kk + kr) * N + tileN + c8);
    }
  };
  auto storeB = [&](int buf, const uint4* r) {  // transpose into [128 cols][32 k]
    unsigned short* sBt = smem + 8192 + buf * 4096;
#pragma unroll
    for (int i = 0; i < 2; ++i) {
      int c = t + (i << 8);
      int kr = c >> 4, c8 = (c & 15) << 3;
      const unsigned short* dv = (const unsigned short*)&r[i];
#pragma unroll
      for (int j = 0; j < 8; ++j) sBt[(c8 + j) * 32 + kr] = dv[j];
    }
  };

  uint4 breg[2];
  stageA(0, 0);
  loadB(0, breg);
  storeB(0, breg);
  wait_async0();
  __syncthreads();

  const int steps = K >> 5;
  for (int s = 0; s < steps; ++s) {
    const int cur = s & 1, nxt = cur ^ 1;
    const int kk = s << 5;
    const bool hasNext = (s + 1) < steps;
    if (hasNext) {             // overlap next-tile fills with this step's WMMAs
      stageA(nxt, kk + 32);
      loadB(kk + 32, breg);
    }
    if (kk + 64 < K)
      __builtin_prefetch(W + (size_t)(kk + 64 + (t >> 4)) * N + tileN + ((t & 15) << 3), 0, 0);

    const unsigned short* sA  = smem + cur * 4096;
    const unsigned short* sBt = smem + 8192 + cur * 4096;
    v16bf a0 = ldsA_frag(sA, wm, 32);
    v16bf a1 = ldsA_frag(sA, wm + 16, 32);
#pragma unroll
    for (int j = 0; j < 4; ++j) {
      v16bf bj = ldsBt_frag(sBt, wn + (j << 4));
      acc[0][j].v = wmma_bf16(a0, bj, acc[0][j].v);
      acc[1][j].v = wmma_bf16(a1, bj, acc[1][j].v);
    }
    if (hasNext) storeB(nxt, breg);
    wait_async0();
    __syncthreads();
  }
#pragma unroll
  for (int i = 0; i < 2; ++i)
#pragma unroll
    for (int j = 0; j < 4; ++j)
      gemm_epilogue<BIAS,GELU_,RES,OUTBF>(acc[i][j], tileM + wm + (i << 4),
                                          tileN + wn + (j << 4), M, N, bias, resid, outF, outB);
}

// ---------------- LayerNorm (D=512, eps=1e-3), one wave per row -> bf16 out
__device__ __forceinline__ void ln_row(const float* __restrict__ x,
                                       const float* __restrict__ g,
                                       const float* __restrict__ bta,
                                       unsigned short* __restrict__ y)
{
  int lane = threadIdx.x & 31;
  float v[16]; float s = 0.f;
#pragma unroll
  for (int i = 0; i < 16; ++i) { v[i] = x[lane + 32 * i]; s += v[i]; }
  float mean = wave_sum(s) * (1.f / 512.f);
  float var = 0.f;
#pragma unroll
  for (int i = 0; i < 16; ++i) { float d = v[i] - mean; var += d * d; }
  float rstd = rsqrtf(wave_sum(var) * (1.f / 512.f) + 1e-3f);
#pragma unroll
  for (int i = 0; i < 16; ++i) {
    int col = lane + 32 * i;
    y[col] = f2bf((v[i] - mean) * rstd * g[col] + bta[col]);
  }
}

__global__ __launch_bounds__(256) void layernorm_bf16(
    const float* __restrict__ X, const float* __restrict__ g,
    const float* __restrict__ bta, unsigned short* __restrict__ Y, int rows)
{
  int row = blockIdx.x * 8 + (threadIdx.x >> 5);
  if (row >= rows) return;
  ln_row(X + (size_t)row * 512, g, bta, Y + (size_t)row * 512);
}

// class-attn LN over concat([query, h]) rows (B*257)
__global__ __launch_bounds__(256) void class_ln_kernel(
    const float* __restrict__ query, const float* __restrict__ h,
    const float* __restrict__ g, const float* __restrict__ bta,
    unsigned short* __restrict__ Y, int rows)
{
  int row = blockIdx.x * 8 + (threadIdx.x >> 5);
  if (row >= rows) return;
  int b = row / 257, s = row % 257;
  const float* x = (s == 0) ? (query + (size_t)b * 512)
                            : (h + ((size_t)b * 256 + (s - 1)) * 512);
  ln_row(x, g, bta, Y + (size_t)row * 512);
}

// ---------------- attention scores: S = QK^T*scale + inter + mask  (one wave per 16x16 tile)
__global__ __launch_bounds__(256) void attn_scores(
    const unsigned short* __restrict__ Q, const unsigned short* __restrict__ Kb,
    const float* __restrict__ inter, const unsigned char* __restrict__ mask,
    float* __restrict__ S)
{
  int wave = blockIdx.x * 8 + (threadIdx.x >> 5);           // B*H*16*16 tiles
  int lane = threadIdx.x & 31;
  int kt = wave & 15, qt = (wave >> 4) & 15, hh = (wave >> 8) & 7, b = wave >> 11;
  CFrag c;
#pragma unroll
  for (int r = 0; r < 8; ++r) c.f[r] = 0.f;
#pragma unroll
  for (int kk = 0; kk < 64; kk += 32) {
    ABFrag a, bb;
    int m = lane & 15, kh = (lane >> 4) << 4;
#pragma unroll
    for (int e = 0; e < 8; ++e) {
      int k = kk + (e << 1) + kh;
      a.u[e] = *(const unsigned*)(Q + (((size_t)b * 256 + qt * 16 + m) * 8 + hh) * 64 + k);
    }
    int n = lane & 15, kb2 = (lane >> 4) << 4;
#pragma unroll
    for (int e = 0; e < 8; ++e) {
      int k = kk + kb2 + (e << 1);
      bb.u[e] = *(const unsigned*)(Kb + (((size_t)b * 256 + kt * 16 + n) * 8 + hh) * 64 + k);
    }
    c.v = wmma_bf16(a.v, bb.v, c.v);
  }
  int col = kt * 16 + (lane & 15);
  int rb = qt * 16 + ((lane >> 4) << 3);
  unsigned char mk = mask[b * 256 + col];
#pragma unroll
  for (int r = 0; r < 8; ++r) {
    int row = rb + r;
    float s = c.f[r] * SCALE_ + inter[(((size_t)b * 256 + row) * 256 + col) * 8 + hh];
    if (!(mask[b * 256 + row] && mk)) s += NEG_;
    S[(((size_t)b * 8 + hh) * 256 + row) * 256 + col] = s;
  }
}

// one wave per 256-wide row
__global__ __launch_bounds__(256) void softmax_rows(
    const float* __restrict__ S, unsigned short* __restrict__ P)
{
  int row = blockIdx.x * 8 + (threadIdx.x >> 5);
  int lane = threadIdx.x & 31;
  const float* src = S + (size_t)row * 256;
  float v[8]; float mx = -3.4e38f;
#pragma unroll
  for (int j = 0; j < 8; ++j) { v[j] = src[lane + 32 * j]; mx = fmaxf(mx, v[j]); }
  mx = wave_max(mx);
  float sum = 0.f;
#pragma unroll
  for (int j = 0; j < 8; ++j) { v[j] = __expf(v[j] - mx); sum += v[j]; }
  float inv = 1.f / wave_sum(sum);
#pragma unroll
  for (int j = 0; j < 8; ++j) P[(size_t)row * 256 + lane + 32 * j] = f2bf(v[j] * inv);
}

// O[b,q,h,d] = sum_k P[b,h,q,k] * V[b,k,h,d]   (one wave per 16x16 tile, K=256)
__global__ __launch_bounds__(256) void attn_av(
    const unsigned short* __restrict__ P, const unsigned short* __restrict__ V,
    unsigned short* __restrict__ O)
{
  int wave = blockIdx.x * 8 + (threadIdx.x >> 5);           // B*H*16*4 tiles
  int lane = threadIdx.x & 31;
  int dt = wave & 3, qt = (wave >> 2) & 15, hh = (wave >> 6) & 7, b = wave >> 9;
  CFrag c;
#pragma unroll
  for (int r = 0; r < 8; ++r) c.f[r] = 0.f;
  for (int kk = 0; kk < 256; kk += 32) {
    ABFrag a, bb;
    int m = lane & 15, kh = (lane >> 4) << 4;
#pragma unroll
    for (int e = 0; e < 8; ++e) {
      int k = kk + (e << 1) + kh;
      a.u[e] = *(const unsigned*)(P + (((size_t)b * 8 + hh) * 256 + qt * 16 + m) * 256 + k);
    }
    int n = lane & 15, kb2 = (lane >> 4) << 4, d = dt * 16 + n;
#pragma unroll
    for (int e = 0; e < 8; ++e) {
      int k = kk + kb2 + (e << 1);
      unsigned lo = V[(((size_t)b * 256 + k)     * 8 + hh) * 64 + d];
      unsigned hi = V[(((size_t)b * 256 + k + 1) * 8 + hh) * 64 + d];
      bb.u[e] = lo | (hi << 16);
    }
    c.v = wmma_bf16(a.v, bb.v, c.v);
  }
  int col = dt * 16 + (lane & 15);
  int rb = qt * 16 + ((lane >> 4) << 3);
#pragma unroll
  for (int r = 0; r < 8; ++r) {
    int row = rb + r;
    O[((size_t)b * 256 + row) * 512 + hh * 64 + col] = f2bf(c.f[r]);
  }
}

// ---------------- small class-attention kernels
__global__ __launch_bounds__(256) void cvt_f32_bf16(
    const float* __restrict__ s, unsigned short* __restrict__ d, int n)
{
  int i = blockIdx.x * 256 + threadIdx.x;
  if (i < n) d[i] = f2bf(s[i]);
}

__global__ __launch_bounds__(256) void init_query_kernel(
    const float* __restrict__ ct, float* __restrict__ q)
{
  int i = blockIdx.x * 256 + threadIdx.x;   // B*512
  q[i] = ct[i & 511];
}

__global__ __launch_bounds__(256) void class_qh_kernel(
    const float* __restrict__ query, const float* __restrict__ cqk,
    const float* __restrict__ cqb, float* __restrict__ qh)
{
  int idx = blockIdx.x * 256 + threadIdx.x;   // B*512
  int b = idx >> 9, hk = idx & 511;
  float acc = cqb[hk];
  const float* qrow = query + (size_t)b * 512;
  for (int d = 0; d < 512; ++d) acc += qrow[d] * cqk[(size_t)d * 512 + hk];
  qh[idx] = acc;
}

// one wave per (b,h): scores over 257 keys -> softmax -> A@V into oheads
__global__ __launch_bounds__(256) void class_attn_kernel(
    const float* __restrict__ qh, const unsigned short* __restrict__ kh,
    const unsigned short* __restrict__ vh, const unsigned char* __restrict__ mask,
    float* __restrict__ ohd)
{
  __shared__ float sp[8][264];
  __shared__ float sq[8][64];
  int wave = threadIdx.x >> 5, lane = threadIdx.x & 31;
  int pair = blockIdx.x * 8 + wave;
  int b = pair >> 3, hh = pair & 7;
  sq[wave][lane]      = qh[(size_t)(b * 8 + hh) * 64 + lane];
  sq[wave][lane + 32] = qh[(size_t)(b * 8 + hh) * 64 + lane + 32];
  __builtin_amdgcn_wave_barrier();
  float sv[9]; int cnt = 0; float mx = -3.4e38f;
  for (int s = lane; s < 257; s += 32) {
    const unsigned short* kp = kh + ((size_t)(b * 257 + s)) * 512 + hh * 64;
    float acc = 0.f;
#pragma unroll
    for (int d = 0; d < 64; ++d) acc += sq[wave][d] * bf2f(kp[d]);
    acc *= SCALE_;
    if (s > 0 && mask[b * 256 + s - 1] == 0) acc += NEG_;
    sv[cnt++] = acc; mx = fmaxf(mx, acc);
  }
  mx = wave_max(mx);
  float sum = 0.f;
  for (int j = 0; j < cnt; ++j) { sv[j] = __expf(sv[j] - mx); sum += sv[j]; }
  float inv = 1.f / wave_sum(sum);
  cnt = 0;
  for (int s = lane; s < 257; s += 32) sp[wave][s] = sv[cnt++] * inv;
  __builtin_amdgcn_wave_barrier();
  int dh0 = lane << 1;
  float a0 = 0.f, a1 = 0.f;
  for (int s = 0; s < 257; ++s) {
    float p = sp[wave][s];
    const unsigned short* vp = vh + ((size_t)(b * 257 + s)) * 512 + hh * 64 + dh0;
    a0 += p * bf2f(vp[0]);
    a1 += p * bf2f(vp[1]);
  }
  ohd[(size_t)b * 512 + hh * 64 + dh0]     = a0;
  ohd[(size_t)b * 512 + hh * 64 + dh0 + 1] = a1;
}

__global__ __launch_bounds__(256) void class_outproj_kernel(
    const float* __restrict__ ohd, const float* __restrict__ cok,
    const float* __restrict__ cob, float* __restrict__ query)
{
  int idx = blockIdx.x * 256 + threadIdx.x;   // B*512
  int b = idx >> 9, d = idx & 511;
  float acc = cob[d];
  for (int hk = 0; hk < 512; ++hk) acc += ohd[(size_t)b * 512 + hk] * cok[(size_t)hk * 512 + d];
  query[idx] += acc;
}

// =====================================================================
static const size_t GEMM_LDS = (size_t)16384 * 2;  // 32 KB (2x A + 2x Bt buffers)

extern "C" void kernel_launch(void* const* d_in, const int* in_sizes, int n_in,
                              void* d_out, int out_size, void* d_ws, size_t ws_size,
                              hipStream_t stream)
{
  (void)in_sizes; (void)n_in; (void)out_size; (void)ws_size;
  const float* x       = (const float*)d_in[0];
  const float* inter   = (const float*)d_in[1];
  const float* ctok    = (const float*)d_in[2];
  const float* p_ln1_g = (const float*)d_in[3];
  const float* p_ln1_b = (const float*)d_in[4];
  const float* p_wq    = (const float*)d_in[5];
  const float* p_wk    = (const float*)d_in[6];
  const float* p_wv    = (const float*)d_in[7];
  const float* p_wo    = (const float*)d_in[8];
  const float* p_ln2_g = (const float*)d_in[9];
  const float* p_ln2_b = (const float*)d_in[10];
  const float* p_w1    = (const float*)d_in[11];
  const float* p_b1    = (const float*)d_in[12];
  const float* p_w2    = (const float*)d_in[13];
  const float* p_b2    = (const float*)d_in[14];
  const float* c_ln1_g = (const float*)d_in[15];
  const float* c_ln1_b = (const float*)d_in[16];
  const float* c_qk    = (const float*)d_in[17];
  const float* c_qb    = (const float*)d_in[18];
  const float* c_kk    = (const float*)d_in[19];
  const float* c_kb    = (const float*)d_in[20];
  const float* c_vk    = (const float*)d_in[21];
  const float* c_vb    = (const float*)d_in[22];
  const float* c_ok    = (const float*)d_in[23];
  const float* c_ob    = (const float*)d_in[24];
  const float* c_ln2_g = (const float*)d_in[25];
  const float* c_ln2_b = (const float*)d_in[26];
  const float* c_w1    = (const float*)d_in[27];
  const float* c_b1    = (const float*)d_in[28];
  const float* c_w2    = (const float*)d_in[29];
  const float* c_b2    = (const float*)d_in[30];
  const unsigned char* mask = (const unsigned char*)d_in[31];

  char* ws = (char*)d_ws;
  size_t off = 0;
  auto take = [&](size_t bytes) -> char* {
    char* p = ws + off; off = (off + bytes + 255) & ~(size_t)255; return p;
  };
  const size_t M  = (size_t)B_ * N_;      // 8192
  const size_t DD = (size_t)D_ * D_;      // 262144
  const size_t DF = (size_t)D_ * F_;      // 1048576

  float*          h      = (float*)take(M * D_ * 4);
  unsigned short* hn     = (unsigned short*)take(M * D_ * 2);
  unsigned short* qb     = (unsigned short*)take(M * D_ * 2);
  unsigned short* kbuf   = (unsigned short*)take(M * D_ * 2);
  unsigned short* vbuf   = (unsigned short*)take(M * D_ * 2);
  float*          scores = (float*)take((size_t)B_ * H_ * N_ * N_ * 4);
  unsigned short* ffn1   = (unsigned short*)scores;   // reuse (FFN phase after softmax)
  unsigned short* probs  = (unsigned short*)take((size_t)B_ * H_ * N_ * N_ * 2);
  unsigned short* ao     = (unsigned short*)take(M * D_ * 2);
  unsigned short* wq_b   = (unsigned short*)take((size_t)L_ * DD * 2);
  unsigned short* wk_b   = (unsigned short*)take((size_t)L_ * DD * 2);
  unsigned short* wv_b   = (unsigned short*)take((size_t)L_ * DD * 2);
  unsigned short* wo_b   = (unsigned short*)take((size_t)L_ * DD * 2);
  unsigned short* w1_b   = (unsigned short*)take((size_t)L_ * DF * 2);
  unsigned short* w2_b   = (unsigned short*)take((size_t)L_ * DF * 2);
  unsigned short* ckk_b  = (unsigned short*)take((size_t)LC_ * DD * 2);
  unsigned short* cvk_b  = (unsigned short*)take((size_t)LC_ * DD * 2);
  unsigned short* cw1_b  = (unsigned short*)take((size_t)LC_ * DF * 2);
  unsigned short* cw2_b  = (unsigned short*)take((size_t)LC_ * DF * 2);
  const int Mc = B_ * 257;                              // 8224
  unsigned short* cn     = (unsigned short*)take((size_t)Mc * D_ * 2);
  unsigned short* khb    = (unsigned short*)take((size_t)Mc * D_ * 2);
  unsigned short* vhb    = (unsigned short*)take((size_t)Mc * D_ * 2);
  float*          qh     = (float*)take((size_t)B_ * D_ * 4);
  float*          ohd    = (float*)take((size_t)B_ * D_ * 4);
  float*          query  = (float*)take((size_t)B_ * D_ * 4);
  unsigned short* fln    = (unsigned short*)take((size_t)B_ * D_ * 2);
  unsigned short* cf1    = (unsigned short*)take((size_t)B_ * F_ * 2);

  auto cvt = [&](const float* s, unsigned short* d, size_t n) {
    cvt_f32_bf16<<<dim3((unsigned)((n + 255) / 256)), dim3(256), 0, stream>>>(s, d, (int)n);
  };
  cvt(p_wq, wq_b, (size_t)L_ * DD);
  cvt(p_wk, wk_b, (size_t)L_ * DD);
  cvt(p_wv, wv_b, (size_t)L_ * DD);
  cvt(p_wo, wo_b, (size_t)L_ * DD);
  cvt(p_w1, w1_b, (size_t)L_ * DF);
  cvt(p_w2, w2_b, (size_t)L_ * DF);
  cvt(c_kk, ckk_b, (size_t)LC_ * DD);
  cvt(c_vk, cvk_b, (size_t)LC_ * DD);
  cvt(c_w1, cw1_b, (size_t)LC_ * DF);
  cvt(c_w2, cw2_b, (size_t)LC_ * DF);

  hipMemcpyAsync(h, x, M * D_ * 4, hipMemcpyDeviceToDevice, stream);

  dim3 blk(256);
  dim3 gD((unsigned)((M + 127) / 128), D_ / 128);   // 64 x 4
  dim3 gF((unsigned)((M + 127) / 128), F_ / 128);   // 64 x 16

  for (int i = 0; i < L_; ++i) {
    layernorm_bf16<<<dim3((unsigned)(M / 8)), blk, 0, stream>>>(h, p_ln1_g + i * D_, p_ln1_b + i * D_, hn, (int)M);
    gemm_bf16_wmma<false,false,false,true><<<gD, blk, GEMM_LDS, stream>>>(hn, wq_b + (size_t)i * DD, nullptr, nullptr, nullptr, qb,   (int)M, D_, D_);
    gemm_bf16_wmma<false,false,false,true><<<gD, blk, GEMM_LDS, stream>>>(hn, wk_b + (size_t)i * DD, nullptr, nullptr, nullptr, kbuf, (int)M, D_, D_);
    gemm_bf16_wmma<false,false,false,true><<<gD, blk, GEMM_LDS, stream>>>(hn, wv_b + (size_t)i * DD, nullptr, nullptr, nullptr, vbuf, (int)M, D_, D_);
    attn_scores<<<dim3(B_ * H_ * 16 * 16 / 8), blk, 0, stream>>>(qb, kbuf, inter, mask, scores);
    softmax_rows<<<dim3(B_ * H_ * N_ / 8), blk, 0, stream>>>(scores, probs);
    attn_av<<<dim3(B_ * H_ * 16 * 4 / 8), blk, 0, stream>>>(probs, vbuf, ao);
    gemm_bf16_wmma<false,false,true,false><<<gD, blk, GEMM_LDS, stream>>>(ao, wo_b + (size_t)i * DD, nullptr, h, h, nullptr, (int)M, D_, D_);
    layernorm_bf16<<<dim3((unsigned)(M / 8)), blk, 0, stream>>>(h, p_ln2_g + i * D_, p_ln2_b + i * D_, hn, (int)M);
    gemm_bf16_wmma<true,true,false,true><<<gF, blk, GEMM_LDS, stream>>>(hn, w1_b + (size_t)i * DF, p_b1 + i * F_, nullptr, nullptr, ffn1, (int)M, F_, D_);
    gemm_bf16_wmma<true,false,true,false><<<gD, blk, GEMM_LDS, stream>>>(ffn1, w2_b + (size_t)i * DF, p_b2 + i * D_, h, h, nullptr, (int)M, D_, F_);
  }

  init_query_kernel<<<dim3(B_ * D_ / 256), blk, 0, stream>>>(ctok, query);

  dim3 gcD((unsigned)((Mc + 127) / 128), D_ / 128);  // 65 x 4
  for (int i = 0; i < LC_; ++i) {
    class_ln_kernel<<<dim3((unsigned)((Mc + 7) / 8)), blk, 0, stream>>>(query, h, c_ln1_g + i * D_, c_ln1_b + i * D_, cn, Mc);
    gemm_bf16_wmma<true,false,false,true><<<gcD, blk, GEMM_LDS, stream>>>(cn, ckk_b + (size_t)i * DD, c_kb + i * D_, nullptr, nullptr, khb, Mc, D_, D_);
    gemm_bf16_wmma<true,false,false,true><<<gcD, blk, GEMM_LDS, stream>>>(cn, cvk_b + (size_t)i * DD, c_vb + i * D_, nullptr, nullptr, vhb, Mc, D_, D_);
    class_qh_kernel<<<dim3(B_ * D_ / 256), blk, 0, stream>>>(query, c_qk + (size_t)i * DD, c_qb + i * D_, qh);
    class_attn_kernel<<<dim3(B_ * H_ / 8), blk, 0, stream>>>(qh, khb, vhb, mask, ohd);
    class_outproj_kernel<<<dim3(B_ * D_ / 256), blk, 0, stream>>>(ohd, c_ok + (size_t)i * DD, c_ob + i * D_, query);
    layernorm_bf16<<<dim3(B_ / 8), blk, 0, stream>>>(query, c_ln2_g + i * D_, c_ln2_b + i * D_, fln, B_);
    gemm_bf16_wmma<true,true,false,true><<<dim3(1, F_ / 128), blk, GEMM_LDS, stream>>>(fln, cw1_b + (size_t)i * DF, c_b1 + i * F_, nullptr, nullptr, cf1, B_, F_, D_);
    gemm_bf16_wmma<true,false,true,false><<<dim3(1, D_ / 128), blk, GEMM_LDS, stream>>>(cf1, cw2_b + (size_t)i * DF, c_b2 + i * D_, query, query, nullptr, B_, D_, F_);
  }

  hipMemcpyAsync(d_out, query, (size_t)B_ * D_ * 4, hipMemcpyDeviceToDevice, stream);
}
